// VAE_RNN_PHY_47931835024020
// MI455X (gfx1250) — matrix operand order repaired and checked by reading, not compiled, not run on hardware
//
#include <hip/hip_runtime.h>
#include <hip/hip_bf16.h>

#define Bb 256
#define Tt 512
#define Hh 512
#define Zz 64
#define Yy 8
#define Uu 8
#define WPB 8   // waves per block (256 threads, wave32)

typedef __bf16 bf16_t;
typedef __attribute__((ext_vector_type(8)))  bf16_t v8bf;
typedef __attribute__((ext_vector_type(16))) bf16_t v16bf;
typedef __attribute__((ext_vector_type(8)))  float  v8f;

__device__ __forceinline__ bf16_t f2bf(float f) {
    union { float f; unsigned u; } v; v.f = f;
    unsigned r = v.u + 0x7FFFu + ((v.u >> 16) & 1u);
    unsigned short h = (unsigned short)(r >> 16);
    return __builtin_bit_cast(bf16_t, h);
}

// ---------------------------------------------------------------------------
// fp32 -> bf16 weight conversion
// ---------------------------------------------------------------------------
__global__ void convert_bf16(const float* __restrict__ src, bf16_t* __restrict__ dst, int n) {
    int i = blockIdx.x * blockDim.x + threadIdx.x;
    if (i < n) dst[i] = f2bf(src[i]);
}

// ---------------------------------------------------------------------------
// init: h = 0 (fp32 + bf16 mirror), loss = 0
// ---------------------------------------------------------------------------
__global__ void init_state(float* h, bf16_t* hbf, float* loss, int n) {
    int i = blockIdx.x * blockDim.x + threadIdx.x;
    if (i < n) { h[i] = 0.0f; hbf[i] = f2bf(0.0f); }
    if (i == 0) *loss = 0.0f;
}

// ---------------------------------------------------------------------------
// First MLP layers (K = 8, below WMMA K): phi_*_1 = relu(x @ W1^T + b1)
// grid(2, B): x selects y/u, y-dim is batch row; block = H threads
// ---------------------------------------------------------------------------
__global__ void first_layers(const float* __restrict__ y, const float* __restrict__ u,
                             const float* __restrict__ Wy1, const float* __restrict__ by1,
                             const float* __restrict__ Wu1, const float* __restrict__ bu1,
                             bf16_t* __restrict__ phi_y1, bf16_t* __restrict__ phi_u1, int t) {
    int n = threadIdx.x;        // 0..H-1
    int b = blockIdx.y;         // 0..B-1
    const float* X; const float* W; const float* bias; bf16_t* out;
    if (blockIdx.x == 0) { X = y; W = Wy1; bias = by1; out = phi_y1; }
    else                 { X = u; W = Wu1; bias = bu1; out = phi_u1; }
    float s = bias[n];
#pragma unroll
    for (int i = 0; i < 8; ++i)
        s += X[(b * 8 + i) * Tt + t] * W[n * 8 + i];
    out[b * Hh + n] = f2bf(fmaxf(s, 0.0f));
}

// ---------------------------------------------------------------------------
// WMMA bf16 GEMM, 2x2 tiles (32x32 output block) per wave:
//   out = act( A[M,K1] @ W[:, :K1]^T (+ A2[M,K2] @ W[:, K1:]^T) + bias )
// W row-major [N, ldW] (PyTorch Linear layout).
// A fragment (16x32 bf16): lane<16 -> M=lane,    K = {k0..k0+7, k0+16..k0+23}
//                          lane>=16 -> M=lane-16, K = {k0+8..k0+15, k0+24..k0+31}
// B fragment (32x16 bf16): lane<16 -> N=lane,    K = {k0..k0+7, k0+16..k0+23}
//                          lane>=16 -> N=lane-16, K = {k0+8..k0+15, k0+24..k0+31}
// C/D (16x16 f32): acc[i] -> row = tileRow + i + (lane<16?0:8), col = tileCol + lane%16
// ---------------------------------------------------------------------------
__device__ __forceinline__ v16bf load_afrag(const bf16_t* __restrict__ A, int ldA,
                                            int rowA, int k0, int aSel) {
    const bf16_t* ap = A + (size_t)rowA * ldA + k0 + aSel;
    v8bf a0 = *(const v8bf*)(ap);
    v8bf a1 = *(const v8bf*)(ap + 16);
    return __builtin_shufflevector(a0, a1, 0,1,2,3,4,5,6,7,8,9,10,11,12,13,14,15);
}

__device__ __forceinline__ v16bf load_bfrag(const bf16_t* __restrict__ W, int ldW,
                                            int colN, int wOff, int k0, int bSel) {
    const bf16_t* bp = W + (size_t)colN * ldW + wOff + k0 + bSel;
    v8bf b0 = *(const v8bf*)(bp);
    v8bf b1 = *(const v8bf*)(bp + 16);
    return __builtin_shufflevector(b0, b1, 0,1,2,3,4,5,6,7,8,9,10,11,12,13,14,15);
}

__device__ __forceinline__ void gemm_seg22(v8f& a00, v8f& a01, v8f& a10, v8f& a11,
                                           const bf16_t* __restrict__ A, int ldA,
                                           const bf16_t* __restrict__ W, int ldW, int wOff,
                                           int Kseg, int rowA0, int rowA1,
                                           int colN0, int colN1, int lane) {
    const int aSel = (lane < 16) ? 0 : 8;
    const int bSel = (lane < 16) ? 0 : 8;
    for (int k0 = 0; k0 < Kseg; k0 += 32) {
        v16bf fa0 = load_afrag(A, ldA, rowA0, k0, aSel);
        v16bf fa1 = load_afrag(A, ldA, rowA1, k0, aSel);
        v16bf fb0 = load_bfrag(W, ldW, colN0, wOff, k0, bSel);
        v16bf fb1 = load_bfrag(W, ldW, colN1, wOff, k0, bSel);
        a00 = __builtin_amdgcn_wmma_f32_16x16x32_bf16(false, fa0, false, fb0, (short)0, a00, false, false);
        a01 = __builtin_amdgcn_wmma_f32_16x16x32_bf16(false, fa0, false, fb1, (short)0, a01, false, false);
        a10 = __builtin_amdgcn_wmma_f32_16x16x32_bf16(false, fa1, false, fb0, (short)0, a10, false, false);
        a11 = __builtin_amdgcn_wmma_f32_16x16x32_bf16(false, fa1, false, fb1, (short)0, a11, false, false);
    }
}

__device__ __forceinline__ void store_tile(const v8f& acc, int rbase, int col, float bcol,
                                           int relu, int N,
                                           float* __restrict__ outF, bf16_t* __restrict__ outB) {
#pragma unroll
    for (int i = 0; i < 8; ++i) {
        float v = acc[i] + bcol;
        if (relu) v = fmaxf(v, 0.0f);
        const int r = rbase + i;
        if (outF) outF[(size_t)r * N + col] = v;
        if (outB) outB[(size_t)r * N + col] = f2bf(v);
    }
}

__global__ void gemm_wmma(const bf16_t* __restrict__ A, int ldA,
                          const bf16_t* __restrict__ A2, int ldA2, int K1, int K2,
                          const bf16_t* __restrict__ W, int ldW,
                          const float* __restrict__ bias,
                          float* __restrict__ outF, bf16_t* __restrict__ outB,
                          int M, int N, int relu) {
    const int lane = threadIdx.x & 31;
    const int wave = threadIdx.x >> 5;
    const int blk  = blockIdx.x * WPB + wave;   // 32x32 output block id
    const int bM = M >> 5;
    const int numBlks = bM * (N >> 5);
    if (blk >= numBlks) return;                 // wave-uniform: EXEC stays all-ones for WMMA
    const int tm = blk % bM;
    const int tn = blk / bM;
    const int rowA0 = tm * 32 + (lane & 15);
    const int rowA1 = rowA0 + 16;
    const int colN0 = tn * 32 + (lane & 15);
    const int colN1 = colN0 + 16;

    v8f a00 = {}, a01 = {}, a10 = {}, a11 = {};
    gemm_seg22(a00, a01, a10, a11, A, ldA, W, ldW, 0, K1, rowA0, rowA1, colN0, colN1, lane);
    if (A2)
        gemm_seg22(a00, a01, a10, a11, A2, ldA2, W, ldW, K1, K2, rowA0, rowA1, colN0, colN1, lane);

    const float b0 = bias ? bias[colN0] : 0.0f;
    const float b1 = bias ? bias[colN1] : 0.0f;
    const int rbase0 = tm * 32 + ((lane < 16) ? 0 : 8);
    const int rbase1 = rbase0 + 16;
    store_tile(a00, rbase0, colN0, b0, relu, N, outF, outB);
    store_tile(a01, rbase0, colN1, b1, relu, N, outF, outB);
    store_tile(a10, rbase1, colN0, b0, relu, N, outF, outB);
    store_tile(a11, rbase1, colN1, b1, relu, N, outF, outB);
}

// ---------------------------------------------------------------------------
// Per-step pointwise: z / decoder / log-prob / KLD / GRU update / loss-accum
// one block per batch row, blockDim = H
// ---------------------------------------------------------------------------
__global__ void step_pointwise(const float* __restrict__ em, const float* __restrict__ elv,
                               const float* __restrict__ pm, const float* __restrict__ plv,
                               const float* __restrict__ gi, const float* __restrict__ gh,
                               const float* __restrict__ eps, const float* __restrict__ y,
                               const float* __restrict__ C,
                               float* __restrict__ h, bf16_t* __restrict__ hbf,
                               float* __restrict__ loss, int t) {
    __shared__ float zs[Zz];
    __shared__ float ss[Zz];
    __shared__ float red[Hh];
    const int b = blockIdx.x;
    const int j = threadIdx.x;
    float part = 0.0f;

    if (j < Zz) {
        float m  = em[b * Zz + j];
        float lv = elv[b * Zz + j];               // relu'd in GEMM epilogue
        float s  = expf(0.5f * lv);
        float e  = eps[((size_t)t * Bb + b) * Zz + j];
        zs[j] = m + s * e;
        ss[j] = s;
        float pv = plv[b * Zz + j];               // relu'd
        float pmv = pm[b * Zz + j];
        part += 0.5f * (pv - lv - 1.0f + (expf(lv) + (m - pmv) * (m - pmv)) / expf(pv));
    }
    __syncthreads();
    if (j < Yy) {
        float dm = 0.0f, zv = 0.0f;
#pragma unroll
        for (int k = 0; k < Zz; ++k) {
            float c = C[j * Zz + k];
            dm += zs[k] * c;
            zv += ss[k] * c;
        }
        float ps = 0.1f + zv;                     // sqrt(0.01) + z_var
        float yt = y[((size_t)b * Yy + j) * Tt + t];
        float d  = (yt - dm) / ps;
        // -logp contribution
        part += 0.5f * d * d + logf(ps) + 0.5f * 1.8378770664093453f; // log(2*pi)
    }

    // GRU update (gates from precomputed gi = phi_u@W_ih^T, gh = h@W_hh^T)
    {
        const size_t base = (size_t)b * (3 * Hh);
        float gr = gi[base + j]            + gh[base + j];
        float gz = gi[base + Hh + j]       + gh[base + Hh + j];
        float gn = gi[base + 2 * Hh + j];
        float hn_ = gh[base + 2 * Hh + j];
        float r  = 1.0f / (1.0f + expf(-gr));
        float zg = 1.0f / (1.0f + expf(-gz));
        float n  = tanhf(gn + r * hn_);
        float ho = h[(size_t)b * Hh + j];
        float hv = (1.0f - zg) * n + zg * ho;
        h[(size_t)b * Hh + j]   = hv;
        hbf[(size_t)b * Hh + j] = f2bf(hv);
    }

    red[j] = part;
    __syncthreads();
    for (int s = Hh / 2; s > 0; s >>= 1) {
        if (j < s) red[j] += red[j + s];
        __syncthreads();
    }
    if (j == 0) atomicAdd(loss, red[0]);
}

// ---------------------------------------------------------------------------
// Host launcher
// ---------------------------------------------------------------------------
extern "C" void kernel_launch(void* const* d_in, const int* in_sizes, int n_in,
                              void* d_out, int out_size, void* d_ws, size_t ws_size,
                              hipStream_t stream) {
    (void)in_sizes; (void)n_in; (void)out_size; (void)ws_size;
    const float* u    = (const float*)d_in[0];
    const float* y    = (const float*)d_in[1];
    const float* eps  = (const float*)d_in[2];
    const float* Wy1  = (const float*)d_in[3];  const float* by1 = (const float*)d_in[4];
    const float* Wy2f = (const float*)d_in[5];  const float* by2 = (const float*)d_in[6];
    const float* Wu1  = (const float*)d_in[7];  const float* bu1 = (const float*)d_in[8];
    const float* Wu2f = (const float*)d_in[9];  const float* bu2 = (const float*)d_in[10];
    const float* We1f = (const float*)d_in[11]; const float* be1 = (const float*)d_in[12];
    const float* We2f = (const float*)d_in[13]; const float* be2 = (const float*)d_in[14];
    const float* Wemf = (const float*)d_in[15]; const float* bem = (const float*)d_in[16];
    const float* Welf = (const float*)d_in[17]; const float* bel = (const float*)d_in[18];
    const float* Wp1f = (const float*)d_in[19]; const float* bp1 = (const float*)d_in[20];
    const float* Wp2f = (const float*)d_in[21]; const float* bp2 = (const float*)d_in[22];
    const float* Wpmf = (const float*)d_in[23]; const float* bpm = (const float*)d_in[24];
    const float* Wplf = (const float*)d_in[25]; const float* bpl = (const float*)d_in[26];
    const float* Wihf = (const float*)d_in[27];
    const float* Whhf = (const float*)d_in[28];
    const float* Cm   = (const float*)d_in[29];
    float* loss = (float*)d_out;

    // ---- workspace carve (256B aligned bump allocator) ----
    char* p = (char*)d_ws;
    auto alloc = [&](size_t bytes) -> void* {
        void* r = (void*)p;
        p += (bytes + 255) & ~(size_t)255;
        return r;
    };
    // bf16 weights
    bf16_t* Wy2 = (bf16_t*)alloc(512 * 512 * 2);
    bf16_t* Wu2 = (bf16_t*)alloc(512 * 512 * 2);
    bf16_t* We1 = (bf16_t*)alloc(512 * 1024 * 2);
    bf16_t* We2 = (bf16_t*)alloc(512 * 512 * 2);
    bf16_t* Wp1 = (bf16_t*)alloc(512 * 512 * 2);
    bf16_t* Wp2 = (bf16_t*)alloc(512 * 512 * 2);
    bf16_t* Wem = (bf16_t*)alloc(64 * 512 * 2);
    bf16_t* Wel = (bf16_t*)alloc(64 * 512 * 2);
    bf16_t* Wpm = (bf16_t*)alloc(64 * 512 * 2);
    bf16_t* Wpl = (bf16_t*)alloc(64 * 512 * 2);
    bf16_t* Wih = (bf16_t*)alloc(1536 * 512 * 2);
    bf16_t* Whh = (bf16_t*)alloc(1536 * 512 * 2);
    // per-step activations (bf16)
    bf16_t* phi_y1 = (bf16_t*)alloc(Bb * Hh * 2);
    bf16_t* phi_u1 = (bf16_t*)alloc(Bb * Hh * 2);
    bf16_t* phi_y  = (bf16_t*)alloc(Bb * Hh * 2);
    bf16_t* phi_u  = (bf16_t*)alloc(Bb * Hh * 2);
    bf16_t* enc1h  = (bf16_t*)alloc(Bb * Hh * 2);
    bf16_t* hp1    = (bf16_t*)alloc(Bb * Hh * 2);
    bf16_t* he     = (bf16_t*)alloc(Bb * Hh * 2);
    bf16_t* hp2    = (bf16_t*)alloc(Bb * Hh * 2);
    bf16_t* hbf    = (bf16_t*)alloc(Bb * Hh * 2);
    // fp32 state / gates / heads
    float* hf  = (float*)alloc(Bb * Hh * 4);
    float* gh  = (float*)alloc((size_t)Bb * 3 * Hh * 4);
    float* gi  = (float*)alloc((size_t)Bb * 3 * Hh * 4);
    float* em  = (float*)alloc(Bb * Zz * 4);
    float* elv = (float*)alloc(Bb * Zz * 4);
    float* pm  = (float*)alloc(Bb * Zz * 4);
    float* plv = (float*)alloc(Bb * Zz * 4);

    // ---- one-time conversions + init ----
    auto conv = [&](const float* s, bf16_t* d, int n) {
        convert_bf16<<<(n + 255) / 256, 256, 0, stream>>>(s, d, n);
    };
    conv(Wy2f, Wy2, 512 * 512);
    conv(Wu2f, Wu2, 512 * 512);
    conv(We1f, We1, 512 * 1024);
    conv(We2f, We2, 512 * 512);
    conv(Wp1f, Wp1, 512 * 512);
    conv(Wp2f, Wp2, 512 * 512);
    conv(Wemf, Wem, 64 * 512);
    conv(Welf, Wel, 64 * 512);
    conv(Wpmf, Wpm, 64 * 512);
    conv(Wplf, Wpl, 64 * 512);
    conv(Wihf, Wih, 1536 * 512);
    conv(Whhf, Whh, 1536 * 512);
    init_state<<<(Bb * Hh + 255) / 256, 256, 0, stream>>>(hf, hbf, loss, Bb * Hh);

    auto gemm = [&](const bf16_t* A, int ldA, const bf16_t* A2, int ldA2, int K1, int K2,
                    const bf16_t* W, int ldW, const float* bias,
                    float* oF, bf16_t* oB, int M, int N, int relu) {
        int blks32 = (M / 32) * (N / 32);               // 32x32 output blocks, one per wave
        int grid = (blks32 + WPB - 1) / WPB;
        gemm_wmma<<<grid, WPB * 32, 0, stream>>>(
            A, ldA, A2, ldA2, K1, K2, W, ldW, bias, oF, oB, M, N, relu);
    };

    // ---- recurrence ----
    for (int t = 0; t < Tt; ++t) {
        first_layers<<<dim3(2, Bb), Hh, 0, stream>>>(y, u, Wy1, by1, Wu1, bu1, phi_y1, phi_u1, t);
        // feature MLP 2nd layers (linear)
        gemm(phi_y1, Hh, nullptr, 0, Hh, 0, Wy2, Hh, by2, nullptr, phi_y, Bb, Hh, 0);
        gemm(phi_u1, Hh, nullptr, 0, Hh, 0, Wu2, Hh, bu2, nullptr, phi_u, Bb, Hh, 0);
        // encoder layer1 on concat(phi_y, h): dual-A GEMM, K = 512 + 512 vs W[512,1024]
        gemm(phi_y, Hh, hbf, Hh, Hh, Hh, We1, 2 * Hh, be1, nullptr, enc1h, Bb, Hh, 1);
        // prior layer1 from h
        gemm(hbf, Hh, nullptr, 0, Hh, 0, Wp1, Hh, bp1, nullptr, hp1, Bb, Hh, 1);
        // GRU gate pre-activations
        gemm(hbf,   Hh, nullptr, 0, Hh, 0, Whh, Hh, nullptr, gh, nullptr, Bb, 3 * Hh, 0);
        gemm(phi_u, Hh, nullptr, 0, Hh, 0, Wih, Hh, nullptr, gi, nullptr, Bb, 3 * Hh, 0);
        // second layers
        gemm(enc1h, Hh, nullptr, 0, Hh, 0, We2, Hh, be2, nullptr, he, Bb, Hh, 1);
        gemm(hp1,   Hh, nullptr, 0, Hh, 0, Wp2, Hh, bp2, nullptr, hp2, Bb, Hh, 1);
        // heads (Z = 64)
        gemm(he,  Hh, nullptr, 0, Hh, 0, Wem, Hh, bem, em,  nullptr, Bb, Zz, 0);
        gemm(he,  Hh, nullptr, 0, Hh, 0, Wel, Hh, bel, elv, nullptr, Bb, Zz, 1);
        gemm(hp2, Hh, nullptr, 0, Hh, 0, Wpm, Hh, bpm, pm,  nullptr, Bb, Zz, 0);
        gemm(hp2, Hh, nullptr, 0, Hh, 0, Wpl, Hh, bpl, plv, nullptr, Bb, Zz, 1);
        // sampling / decoder / loss / GRU state update
        step_pointwise<<<Bb, Hh, 0, stream>>>(em, elv, pm, plv, gi, gh, eps, y, Cm,
                                              hf, hbf, loss, t);
    }
}
